// MultiHeadAttention_PE_43155831390693
// MI455X (gfx1250) — matrix operand order repaired and checked by reading, not compiled, not run on hardware
//
#include <hip/hip_runtime.h>
#include <hip/hip_bf16.h>
#include <stdint.h>

// ---------------------------------------------------------------------------
// MI455X (gfx1250) multi-head attention, bf16 WMMA path, fp32 accumulate.
// ~52 GFLOP vs ~55 MB HBM traffic -> compute bound -> everything on
// v_wmma_f32_16x16x32_bf16.  K tiles stream memory->LDS via
// global_load_async_to_lds_b32 (ASYNCcnt, double buffered); V is transposed
// through VGPRs into LDS so every WMMA fragment read is contiguous b128.
// Workspace: Q,K,V bf16 [B,H,S,64] + U bf16 [B,S,512] = 32 MB.
// ---------------------------------------------------------------------------

typedef __attribute__((ext_vector_type(16))) __bf16 v16bf;
typedef __attribute__((ext_vector_type(8)))  float  v8f;

#define DEV __device__ __forceinline__

DEV v8f wmma_bf16(v16bf a, v16bf b, v8f c) {
  // (neg_a, A, neg_b, B, c_mod, C, reuse_a, reuse_b)
  return __builtin_amdgcn_wmma_f32_16x16x32_bf16(false, a, false, b, (short)0, c,
                                                 false, false);
}

DEV uint32_t pack2bf(float a, float b) {
  union { __bf16 h[2]; uint32_t w; } u;
  u.h[0] = (__bf16)a;
  u.h[1] = (__bf16)b;
  return u.w;
}

// Async global->LDS copy, 4B per lane (GVS mode: SGPR64 base + VGPR32 byte
// offset).  VDST = LDS byte address (low 32 bits of the flat shared pointer).
// Tracked by ASYNCcnt.
DEV void async_copy_b32(uint32_t lds_addr, uint64_t gbase, uint32_t goff) {
  asm volatile("global_load_async_to_lds_b32 %0, %1, %2 offset:0"
               :
               : "v"(lds_addr), "v"(goff), "s"(gbase)
               : "memory");
}

// A fragment (16x32 bf16), rows m0.., cols k0.. of a row-major LDS tile.
// Contiguous k order (B uses the same): lane L holds row L%16; K = j + 16*(L/16).
// One contiguous 32B run per lane -> 2x ds_load_b128.
DEV v16bf load_a_frag(const __bf16* p, int m0, int k0, int ld) {
  const int lane = threadIdx.x & 31;
  const int m = m0 + (lane & 15);
  const int kb = k0 + ((lane >> 4) << 4);
  v16bf a;
#pragma unroll
  for (int j = 0; j < 16; ++j) {
    int k = kb + j;
    a[j] = p[m * ld + k];
  }
  return a;
}

// B fragment (32x16 bf16) where B[K][N] = p[(n0+N)*ld + k0+K]  (N-major tile:
// the W[n][k] case, the K-cache case, and transposed-V case).  One contiguous
// 32B run per lane -> 2x ds_load_b128.
DEV v16bf load_b_frag_nk(const __bf16* p, int k0, int n0, int ld) {
  const int lane = threadIdx.x & 31;
  const int n = n0 + (lane & 15);
  const int kb = k0 + ((lane >> 4) << 4);
  v16bf b;
#pragma unroll
  for (int j = 0; j < 16; ++j) b[j] = p[n * ld + kb + j];
  return b;
}

// ---------------------------------------------------------------------------
// GEMM: out[m,n] = sum_k in[m,k] * W[n,k] + bias[n],  K = N = 512.
// 64x128 output tile per 256-thread block; 8 waves (4 M-strips x 2 N-halves),
// each wave: one A fragment reused across 4 WMMAs per 32-deep K step.
// IN_BF16:  input already bf16 (attention output U), else fp32->bf16 on load.
// HEADS_OUT: scatter bf16 result to [B,H,S,64]; else store fp32 [M,512].
// ---------------------------------------------------------------------------
template <bool IN_BF16, bool HEADS_OUT>
__global__ __launch_bounds__(256) void gemm512_bias(
    const void* __restrict__ inv, const float* __restrict__ W,
    const float* __restrict__ bias, void* __restrict__ outv, int S) {
  constexpr int KD = 512;
  __shared__ uint32_t As32[64 * 16];   // 64  x 32 bf16 input tile
  __shared__ uint32_t Ws32[128 * 16];  // 128(n) x 32(k) bf16 weight tile
  __bf16* As = (__bf16*)As32;
  __bf16* Ws = (__bf16*)Ws32;

  const int mt = blockIdx.x * 64;
  const int nt = blockIdx.y * 128;
  const int t = threadIdx.x;
  const int wave = t >> 5;
  const int lane = t & 31;
  const int m0 = (wave & 3) * 16;
  const int n0 = (wave >> 2) * 64;

  v8f acc[4] = {{}, {}, {}, {}};

  for (int k0 = 0; k0 < KD; k0 += 32) {
    __syncthreads();
    if constexpr (IN_BF16) {
      const uint32_t* src32 = (const uint32_t*)inv;
      for (int i = t; i < 1024; i += 256) {
        int r = i >> 4, c = i & 15;
        As32[i] = src32[(size_t)(mt + r) * (KD / 2) + (k0 >> 1) + c];
      }
    } else {
      const float2* src = (const float2*)inv;
      for (int i = t; i < 1024; i += 256) {
        int r = i >> 4, c = i & 15;
        float2 v = src[(size_t)(mt + r) * (KD / 2) + (k0 >> 1) + c];
        As32[i] = pack2bf(v.x, v.y);
      }
    }
    {
      const float2* src = (const float2*)W;
      for (int i = t; i < 2048; i += 256) {
        int r = i >> 4, c = i & 15;
        float2 v = src[(size_t)(nt + r) * (KD / 2) + (k0 >> 1) + c];
        Ws32[i] = pack2bf(v.x, v.y);
      }
    }
    __syncthreads();

    v16bf a = load_a_frag(As, m0, 0, 32);
#pragma unroll
    for (int j = 0; j < 4; ++j)
      acc[j] = wmma_bf16(a, load_b_frag_nk(Ws, 0, n0 + 16 * j, 32), acc[j]);
  }

  // Epilogue: C layout -> memory.  M = r + 8*(lane/16), N = lane%16.
  const int nc = lane & 15;
#pragma unroll
  for (int r = 0; r < 8; ++r) {
    const int m = mt + m0 + r + ((lane >> 4) << 3);
#pragma unroll
    for (int j = 0; j < 4; ++j) {
      const int n = nt + n0 + 16 * j + nc;
      const float v = acc[j][r] + bias[n];
      if constexpr (HEADS_OUT) {
        __bf16* out = (__bf16*)outv;
        const int b = m / S, s = m % S;
        out[((size_t)(b * 8 + (n >> 6)) * S + s) * 64 + (n & 63)] = (__bf16)v;
      } else {
        ((float*)outv)[(size_t)m * 512 + n] = v;
      }
    }
  }
}

// ---------------------------------------------------------------------------
// Flash attention: one (b, h, 64-q-row) tile per 128-thread block (4 waves).
// Each wave owns 16 q rows; K tiles stream in via async global->LDS loads
// (double buffered, ASYNCcnt); V tiles are transposed through VGPRs into
// [d][key] LDS; online softmax; P round-trips LDS (C-layout -> A-layout).
// Q,K,V: bf16 [B*H, S, 64];  U: bf16 [B, S, 512] (heads re-concatenated).
// ---------------------------------------------------------------------------
__global__ __launch_bounds__(128) void attn64(const __bf16* __restrict__ Q,
                                              const __bf16* __restrict__ K,
                                              const __bf16* __restrict__ V,
                                              __bf16* __restrict__ U) {
  constexpr int S = 2048;
  constexpr int NT = S / 64;              // 32 kv tiles
  __shared__ uint32_t Qs32[64 * 32];      // 64 q   x 64 d   bf16
  __shared__ uint32_t Ks32[2][64 * 32];   // double-buffered 64 key x 64 d
  __shared__ uint32_t Vt32[64 * 32];      // 64 d   x 64 key bf16 (transposed)
  __shared__ uint32_t Ps32[4 * 16 * 32];  // per-wave 16 q x 64 key bf16
  __bf16* Qs = (__bf16*)Qs32;
  __bf16* Vt = (__bf16*)Vt32;

  const int t = threadIdx.x;
  const int wave = t >> 5;
  const int lane = t & 31;
  const int bh = blockIdx.x;  // b*8 + h
  const int qt = blockIdx.y;  // q tile of 64 rows
  const int b = bh >> 3, h = bh & 7;

  const uint32_t* Qg32 = (const uint32_t*)(Q + ((size_t)bh * S + qt * 64) * 64);
  const uint64_t Kbase = (uint64_t)(uintptr_t)(K + (size_t)bh * S * 64);
  const uint32_t* Vg32 = (const uint32_t*)(V + (size_t)bh * S * 64);

  for (int i = t; i < 2048; i += 128) Qs32[i] = Qg32[i];

  // Prologue: async-load K tile 0 into buffer 0 (16 async b32 per wave).
  for (int i = t; i < 2048; i += 128)
    async_copy_b32((uint32_t)(uintptr_t)&Ks32[0][i], Kbase, (uint32_t)(i * 4));

  v8f o0 = {}, o1 = {}, o2 = {}, o3 = {};
  float mrow[8], lrow[8];
#pragma unroll
  for (int r = 0; r < 8; ++r) { mrow[r] = -1e30f; lrow[r] = 0.0f; }

  __bf16* Pw = ((__bf16*)Ps32) + wave * (16 * 64);
  const int qb = (lane >> 4) << 3;  // row base of this half-wave in C layout
  const int nc = lane & 15;

  for (int kt = 0; kt < NT; ++kt) {
    __syncthreads();  // previous iteration's LDS reads done
    // V tile: transpose into [d][key].  Lane-consecutive keys within a d-row
    // keep the LDS b16 stores conflict-free; the strided global reads hit L2
    // (each 8KB tile is reused by all 32 q-blocks of this head).
    for (int i = t; i < 2048; i += 128) {
      int dp = i >> 6;    // d pair 0..31
      int key = i & 63;
      union { uint32_t w; __bf16 h[2]; } u;
      u.w = Vg32[(size_t)kt * 2048 + key * 32 + dp];
      Vt[(2 * dp) * 64 + key] = u.h[0];
      Vt[(2 * dp + 1) * 64 + key] = u.h[1];
    }
    // Issue next K tile into the other buffer (overlaps with this tile's
    // compute), then wait for the current tile: async loads complete in
    // order, so asynccnt <= 16 means tile kt has fully landed.
    if (kt + 1 < NT) {
      const uint32_t gb = (uint32_t)((kt + 1) * 8192);
      for (int i = t; i < 2048; i += 128)
        async_copy_b32((uint32_t)(uintptr_t)&Ks32[(kt + 1) & 1][i], Kbase,
                       gb + (uint32_t)(i * 4));
      asm volatile("s_wait_asynccnt 0x10" ::: "memory");
    } else {
      asm volatile("s_wait_asynccnt 0x0" ::: "memory");
    }
    __syncthreads();

    const __bf16* Ks = (const __bf16*)Ks32[kt & 1];

    // S = Q * K^T  (16 q rows x 64 keys per wave)
    v8f s0 = {}, s1 = {}, s2 = {}, s3 = {};
#pragma unroll
    for (int kk = 0; kk < 64; kk += 32) {
      v16bf a = load_a_frag(Qs, wave * 16, kk, 64);
      s0 = wmma_bf16(a, load_b_frag_nk(Ks, kk, 0, 64), s0);
      s1 = wmma_bf16(a, load_b_frag_nk(Ks, kk, 16, 64), s1);
      s2 = wmma_bf16(a, load_b_frag_nk(Ks, kk, 32, 64), s2);
      s3 = wmma_bf16(a, load_b_frag_nk(Ks, kk, 48, 64), s3);
    }

    // Online softmax. Row m = r + 8*(lane/16); all 16 N live in the 16 lanes
    // of this half-wave, so row reductions are xor-shuffles over masks 1..8.
    const float sc = 0.125f;  // 1/sqrt(64)
#pragma unroll
    for (int r = 0; r < 8; ++r) {
      float e0 = s0[r] * sc, e1 = s1[r] * sc, e2 = s2[r] * sc, e3 = s3[r] * sc;
      float tm = fmaxf(fmaxf(e0, e1), fmaxf(e2, e3));
#pragma unroll
      for (int off = 1; off < 16; off <<= 1)
        tm = fmaxf(tm, __shfl_xor(tm, off, 32));
      const float mn = fmaxf(mrow[r], tm);
      const float p0 = __expf(e0 - mn), p1 = __expf(e1 - mn);
      const float p2 = __expf(e2 - mn), p3 = __expf(e3 - mn);
      __bf16* pr = Pw + (qb + r) * 64 + nc;
      pr[0] = (__bf16)p0;
      pr[16] = (__bf16)p1;
      pr[32] = (__bf16)p2;
      pr[48] = (__bf16)p3;
      float ts = (p0 + p1) + (p2 + p3);
#pragma unroll
      for (int off = 1; off < 16; off <<= 1) ts += __shfl_xor(ts, off, 32);
      const float al = __expf(mrow[r] - mn);
      lrow[r] = lrow[r] * al + ts;
      mrow[r] = mn;
      o0[r] *= al; o1[r] *= al; o2[r] *= al; o3[r] *= al;
    }

    // Same-wave LDS RAW: DS ops are in-order per wave; wait for the P stores.
    asm volatile("s_wait_dscnt 0" ::: "memory");

    // O += P * V.  B[K=key][N=d] = Vt[d*64 + key] -> contiguous nk loads.
#pragma unroll
    for (int kk = 0; kk < 64; kk += 32) {
      v16bf pa = load_a_frag(Pw, 0, kk, 64);
      o0 = wmma_bf16(pa, load_b_frag_nk(Vt, kk, 0, 64), o0);
      o1 = wmma_bf16(pa, load_b_frag_nk(Vt, kk, 16, 64), o1);
      o2 = wmma_bf16(pa, load_b_frag_nk(Vt, kk, 32, 64), o2);
      o3 = wmma_bf16(pa, load_b_frag_nk(Vt, kk, 48, 64), o3);
    }
  }

  // Normalize and write bf16 into the concatenated [B,S,512] buffer.
#pragma unroll
  for (int r = 0; r < 8; ++r) {
    const float inv = 1.0f / lrow[r];
    const int row = qt * 64 + wave * 16 + qb + r;
    __bf16* dst = U + (size_t)(b * S + row) * 512 + h * 64 + nc;
    dst[0]  = (__bf16)(o0[r] * inv);
    dst[16] = (__bf16)(o1[r] * inv);
    dst[32] = (__bf16)(o2[r] * inv);
    dst[48] = (__bf16)(o3[r] * inv);
  }
}

// ---------------------------------------------------------------------------
extern "C" void kernel_launch(void* const* d_in, const int* in_sizes, int n_in,
                              void* d_out, int out_size, void* d_ws,
                              size_t ws_size, hipStream_t stream) {
  (void)in_sizes; (void)n_in; (void)out_size; (void)ws_size;
  const float* x  = (const float*)d_in[0];
  const float* q  = (const float*)d_in[1];
  const float* Wq = (const float*)d_in[2];
  const float* bq = (const float*)d_in[3];
  const float* Wk = (const float*)d_in[4];
  const float* bk = (const float*)d_in[5];
  const float* Wv = (const float*)d_in[6];
  const float* bv = (const float*)d_in[7];
  const float* Wo = (const float*)d_in[8];
  const float* bo = (const float*)d_in[9];
  float* out = (float*)d_out;

  const int B = 4, S = 2048, D = 512;
  const size_t elems = (size_t)B * S * D;  // 4,194,304
  __bf16* Qb = (__bf16*)d_ws;              // [B,H,S,64] bf16
  __bf16* Kb = Qb + elems;
  __bf16* Vb = Kb + elems;
  __bf16* Ub = Vb + elems;                 // [B,S,512] bf16
  // total ws use: 4 * 4Mi * 2B = 32 MB

  dim3 gproj(B * S / 64, D / 128);  // 128 x 4
  gemm512_bias<false, true><<<gproj, 256, 0, stream>>>(q, Wq, bq, Qb, S);
  gemm512_bias<false, true><<<gproj, 256, 0, stream>>>(x, Wk, bk, Kb, S);
  gemm512_bias<false, true><<<gproj, 256, 0, stream>>>(x, Wv, bv, Vb, S);

  dim3 gattn(B * 8, S / 64);  // 32 x 32
  attn64<<<gattn, 128, 0, stream>>>(Qb, Kb, Vb, Ub);

  gemm512_bias<true, false><<<gproj, 256, 0, stream>>>(Ub, Wo, bo, out, S);
}
